// Encoder_67216238182498
// MI455X (gfx1250) — matrix operand order repaired and checked by reading, not compiled, not run on hardware
//
#include <hip/hip_runtime.h>

typedef __attribute__((ext_vector_type(16))) _Float16 v16h;
typedef __attribute__((ext_vector_type(8)))  float    v8f;

#define B_  128
#define T_  1024
#define F_  64
#define H_  128
#define G4  512   // 4*H

union AB16 { v16h h; uint4 u[2]; };

__device__ __forceinline__ float sigmoidf_(float x) { return 1.0f / (1.0f + __expf(-x)); }
__device__ __forceinline__ float tanhf_(float x)    { return 2.0f / (1.0f + __expf(-2.0f * x)) - 1.0f; }

// ---------------------------------------------------------------- f32 -> f16
__global__ void cvt_f16_kernel(const float* __restrict__ in, _Float16* __restrict__ out, int n) {
    int i = blockIdx.x * blockDim.x + threadIdx.x;
    int stride = gridDim.x * blockDim.x;
    for (; i < n; i += stride) out[i] = (_Float16)in[i];
}

// ---- pack W[K,512] (f32) into WMMA B-operand layout: dword = {K=2j+1, K=2j} f16
// out dword index = ((kt*32 + nt)*32 + lane)*8 + j ; lanes 16-31 hold K=16+2j
__global__ void pack_b_kernel(const float* __restrict__ W, unsigned int* __restrict__ out, int ndw) {
    int idx = blockIdx.x * blockDim.x + threadIdx.x;
    if (idx >= ndw) return;
    int j    = idx & 7;
    int lane = (idx >> 3) & 31;
    int nt   = (idx >> 8) & 31;
    int kt   = idx >> 13;
    int k0 = kt * 32 + ((lane & 16) ? 16 : 0) + 2 * j;
    int n  = nt * 16 + (lane & 15);
    _Float16 lo = (_Float16)W[(size_t)k0 * G4 + n];
    _Float16 hi = (_Float16)W[(size_t)(k0 + 1) * G4 + n];
    unsigned short uslo, ushi;
    __builtin_memcpy(&uslo, &lo, 2);
    __builtin_memcpy(&ushi, &hi, 2);
    out[idx] = ((unsigned int)ushi << 16) | (unsigned int)uslo;
}

// ---------------- xz = A[M,K](f16) @ Wpk + bias -> f16 [M,512], M = B*T = 131072
// one wave per 16x16 output tile; grid (8192, 4), 8 waves/block
__global__ __launch_bounds__(256) void gemm_xz_kernel(
    const _Float16* __restrict__ A, int lda,
    const uint4* __restrict__ Bpk, const float* __restrict__ bias,
    _Float16* __restrict__ xz, int ktiles)
{
    int lane = threadIdx.x & 31;
    int w    = threadIdx.x >> 5;
    int mt   = blockIdx.x;
    int nt   = blockIdx.y * 8 + w;
    int m0   = mt * 16;
    int arow = m0 + (lane & 15);
    int koff = (lane & 16) ? 16 : 0;

    v8f acc = {};
    for (int kt = 0; kt < ktiles; ++kt) {
        int kbase = kt * 32;
        AB16 a, b;
        a.u[0] = *(const uint4*)&A[(size_t)arow * lda + kbase + koff];
        a.u[1] = *(const uint4*)&A[(size_t)arow * lda + kbase + koff + 8];
        const uint4* bp = &Bpk[(size_t)((kt * 32 + nt) * 32 + lane) * 2];
        b.u[0] = bp[0];
        b.u[1] = bp[1];
        acc = __builtin_amdgcn_wmma_f32_16x16x32_f16(false, a.h, false, b.h,
                                                     (short)0, acc, false, false);
    }
    int n     = nt * 16 + (lane & 15);
    float bv  = bias[n];
    int mbase = m0 + ((lane & 16) ? 8 : 0);
    for (int r = 0; r < 8; ++r)
        xz[(size_t)(mbase + r) * G4 + n] = (_Float16)(acc[r] + bv);
}

// ---------------- persistent single-workgroup LSTM recurrence over T steps.
// 1 block x 512 threads (16 waves). Each wave owns 4 (m-tile, h-tile) positions
// sharing one h-tile, so its 16 U-operand tiles (4 gates x 4 k-steps) are
// register-resident for the whole kernel. h ping-pongs in LDS (2 x 32KB), one
// barrier per timestep. c stays in registers (fixed ownership).
__global__ __launch_bounds__(512) void lstm_seq_kernel(
    const _Float16* __restrict__ xz,      // [B,T,512] f16
    const uint4*    __restrict__ Upk,     // packed B-layout, 4 k-tiles
    const float*    __restrict__ h_init,  // [B,H] or nullptr (zeros)
    const float*    __restrict__ c_init,  // [B,H] or nullptr (zeros)
    _Float16*       __restrict__ hs_out,  // [B,T,H] f16 or nullptr
    float*          __restrict__ h_final, // [B,H]
    float*          __restrict__ c_final) // [B,H]
{
    extern __shared__ char smem[];
    _Float16* hbuf0 = (_Float16*)smem;              // 32KB
    _Float16* hbuf1 = (_Float16*)(smem + 32768);    // 32KB

    int tid  = threadIdx.x;
    int lane = tid & 31;
    int w    = tid >> 5;          // 0..15
    int ht   = w & 7;             // h-column tile, fixed per wave
    int mtg  = w >> 3;            // 0..1 -> m-tiles mtg*4 .. mtg*4+3
    int hc0  = ht * 16;
    int msub   = (lane & 16) ? 8 : 0;
    int nlocal = lane & 15;
    int koff   = (lane & 16) ? 16 : 0;
    int arowl  = lane & 15;

    // U tiles register-resident for the whole recurrence: breg[gate][kt]
    AB16 breg[4][4];
#pragma unroll
    for (int g = 0; g < 4; ++g) {
        int ntg = g * 8 + ht;
#pragma unroll
        for (int kt = 0; kt < 4; ++kt) {
            const uint4* bp = &Upk[(size_t)((kt * 32 + ntg) * 32 + lane) * 2];
            breg[g][kt].u[0] = bp[0];
            breg[g][kt].u[1] = bp[1];
        }
    }

    // initialize h(t=-1) into hbuf0
    for (int i = tid; i < B_ * H_; i += 512)
        hbuf0[i] = h_init ? (_Float16)h_init[i] : (_Float16)0.0f;

    float creg[4][8];
#pragma unroll
    for (int p = 0; p < 4; ++p) {
        int m0 = (mtg * 4 + p) * 16;
#pragma unroll
        for (int r = 0; r < 8; ++r) {
            int b = m0 + msub + r;
            creg[p][r] = c_init ? c_init[b * H_ + hc0 + nlocal] : 0.0f;
        }
    }
    __syncthreads();

    for (int t = 0; t < T_; ++t) {
        const _Float16* rbuf = (t & 1) ? hbuf1 : hbuf0;   // h(t-1)
        _Float16*       wbuf = (t & 1) ? hbuf0 : hbuf1;   // h(t)

#pragma unroll
        for (int p = 0; p < 4; ++p) {
            int m0   = (mtg * 4 + p) * 16;
            int arow = m0 + arowl;

            // prefetch next timestep's xz slice for this position
            if (t + 1 < T_) {
                size_t pre = ((size_t)(m0 + msub) * T_ + (t + 1)) * G4 + hc0 + nlocal;
                __builtin_prefetch(&xz[pre], 0, 1);
            }

            AB16 a[4];
#pragma unroll
            for (int kt = 0; kt < 4; ++kt) {
                int kbase = kt * 32;
                a[kt].u[0] = *(const uint4*)&rbuf[arow * H_ + kbase + koff];
                a[kt].u[1] = *(const uint4*)&rbuf[arow * H_ + kbase + koff + 8];
            }

            v8f acc[4];
#pragma unroll
            for (int g = 0; g < 4; ++g) {
                v8f cacc = {};
#pragma unroll
                for (int kt = 0; kt < 4; ++kt)
                    cacc = __builtin_amdgcn_wmma_f32_16x16x32_f16(false, a[kt].h,
                                                                  false, breg[g][kt].h,
                                                                  (short)0, cacc, false, false);
                acc[g] = cacc;
            }

#pragma unroll
            for (int r = 0; r < 8; ++r) {
                int b = m0 + msub + r;
                size_t base = ((size_t)b * T_ + t) * G4 + hc0 + nlocal;
                float zi = acc[0][r] + (float)xz[base + 0 * H_];
                float zf = acc[1][r] + (float)xz[base + 1 * H_];
                float zg = acc[2][r] + (float)xz[base + 2 * H_];
                float zo = acc[3][r] + (float)xz[base + 3 * H_];
                float ig = sigmoidf_(zi);
                float fg = sigmoidf_(zf);
                float gg = tanhf_(zg);
                float og = sigmoidf_(zo);
                float cv = fg * creg[p][r] + ig * gg;
                creg[p][r] = cv;
                float hv = og * tanhf_(cv);
                wbuf[b * H_ + hc0 + nlocal] = (_Float16)hv;
                if (hs_out)
                    hs_out[((size_t)b * T_ + t) * H_ + hc0 + nlocal] = (_Float16)hv;
            }
        }
        __syncthreads();   // h(t) complete before anyone reads it at t+1
    }

    // T_ is even -> final h lives in hbuf0
#pragma unroll
    for (int p = 0; p < 4; ++p) {
        int m0 = (mtg * 4 + p) * 16;
#pragma unroll
        for (int r = 0; r < 8; ++r) {
            int b = m0 + msub + r;
            c_final[b * H_ + hc0 + nlocal] = creg[p][r];
            h_final[b * H_ + hc0 + nlocal] = (float)hbuf0[b * H_ + hc0 + nlocal];
        }
    }
}

extern "C" void kernel_launch(void* const* d_in, const int* in_sizes, int n_in,
                              void* d_out, int out_size, void* d_ws, size_t ws_size,
                              hipStream_t stream) {
    const float* x  = (const float*)d_in[0];  // [B,T,F]
    const float* W0 = (const float*)d_in[1];  // [F,512]
    const float* U0 = (const float*)d_in[2];  // [H,512]
    const float* b0 = (const float*)d_in[3];
    const float* W1 = (const float*)d_in[4];  // [H,512]
    const float* U1 = (const float*)d_in[5];
    const float* b1 = (const float*)d_in[6];

    char* ws = (char*)d_ws;
    size_t off = 0;
    auto alloc = [&](size_t bytes) { char* p = ws + off; off = (off + bytes + 255) & ~(size_t)255; return p; };

    _Float16*     xzbuf = (_Float16*)alloc((size_t)B_ * T_ * G4 * 2);   // 128MB, reused
    _Float16*     hs0   = (_Float16*)alloc((size_t)B_ * T_ * H_ * 2);   // 32MB
    _Float16*     xf16  = (_Float16*)alloc((size_t)B_ * T_ * F_ * 2);   // 16MB
    unsigned int* W0pk  = (unsigned int*)alloc(2 * 8192 * 4);           // K=64  -> 2 k-tiles
    unsigned int* U0pk  = (unsigned int*)alloc(4 * 8192 * 4);           // K=128 -> 4 k-tiles
    unsigned int* W1pk  = (unsigned int*)alloc(4 * 8192 * 4);
    unsigned int* U1pk  = (unsigned int*)alloc(4 * 8192 * 4);
    float*        h0f   = (float*)alloc((size_t)B_ * H_ * 4);
    float*        c0f   = (float*)alloc((size_t)B_ * H_ * 4);

    // convert input activations to f16
    cvt_f16_kernel<<<2048, 256, 0, stream>>>(x, xf16, B_ * T_ * F_);

    // pack weights into WMMA B-operand layout
    pack_b_kernel<<<(2 * 8192 + 255) / 256, 256, 0, stream>>>(W0, W0pk, 2 * 8192);
    pack_b_kernel<<<(4 * 8192 + 255) / 256, 256, 0, stream>>>(U0, U0pk, 4 * 8192);
    pack_b_kernel<<<(4 * 8192 + 255) / 256, 256, 0, stream>>>(W1, W1pk, 4 * 8192);
    pack_b_kernel<<<(4 * 8192 + 255) / 256, 256, 0, stream>>>(U1, U1pk, 4 * 8192);

    const size_t lds_bytes = 2 * (size_t)B_ * H_ * 2;  // 64KB h ping-pong

    // layer 0
    gemm_xz_kernel<<<dim3(8192, 4), 256, 0, stream>>>(xf16, F_, (const uint4*)W0pk, b0, xzbuf, 2);
    lstm_seq_kernel<<<1, 512, lds_bytes, stream>>>(xzbuf, (const uint4*)U0pk,
                                                   nullptr, nullptr, hs0, h0f, c0f);
    // layer 1 (initial state = layer 0 final state)
    gemm_xz_kernel<<<dim3(8192, 4), 256, 0, stream>>>(hs0, H_, (const uint4*)W1pk, b1, xzbuf, 4);
    lstm_seq_kernel<<<1, 512, lds_bytes, stream>>>(xzbuf, (const uint4*)U1pk,
                                                   h0f, c0f, nullptr,
                                                   (float*)d_out, (float*)d_out + B_ * H_);
    (void)in_sizes; (void)n_in; (void)out_size; (void)ws_size;
}
